// DagEncoder_43645457662072
// MI455X (gfx1250) — compile-verified, matrix operand order checked by reading, not run on hardware
//
#include <hip/hip_runtime.h>

#define NUM_NODES 2000000
#define NUM_DAGS  20000
#define F_DIM 5
#define E_DIM 32
#define H_DIM 64
#define DIN   37   // F + E
#define DINP  64   // padded K

typedef __attribute__((ext_vector_type(16))) _Float16 v16h;
typedef __attribute__((ext_vector_type(8)))  _Float16 v8h;
typedef __attribute__((ext_vector_type(4)))  _Float16 v4h;
typedef __attribute__((ext_vector_type(8)))  float    v8f;

// Build a 16x16x32 WMMA A/B fragment from an LDS row of 64 f16 (K-major,
// 16B aligned). ISA layout (CDNA5 ISA): lanes 0-15 hold
// K = {kc*32+0..7, kc*32+16..23}; lanes 16-31 hold K = {kc*32+8..15, kc*32+24..31}.
// Both 8-element groups are contiguous -> two ds_load_b128.
__device__ __forceinline__ v16h load_frag(const _Float16* rowp, int kc, int hl) {
    const _Float16* p = rowp + kc * 32 + hl * 8;
    v8h lo = *(const v8h*)(p);
    v8h hi = *(const v8h*)(p + 16);
    return __builtin_shufflevector(lo, hi, 0,1,2,3,4,5,6,7,8,9,10,11,12,13,14,15);
}

// K-layout permutation: concat is stored as [h_node(0..31) | x(32..36) | pad].
// W1 rows are permuted identically when transposed into LDS, so the GEMM is
// mathematically identical to concat([x, h_node]) @ W1.

__global__ void __launch_bounds__(128)
dag_mlp_segsum(const float* __restrict__ h_node, const float* __restrict__ x,
               const float* __restrict__ W1, const float* __restrict__ b1,
               const float* __restrict__ W2, const float* __restrict__ b2,
               const float* __restrict__ W3, const float* __restrict__ b3,
               const int* __restrict__ ptr, float* __restrict__ out) {
    // Weights transposed to [out][in] so B fragments are K-contiguous in LDS.
    __shared__ __align__(16) _Float16 Wt1[64 * 64];   // [o][k], K-permuted + padded
    __shared__ __align__(16) _Float16 Wt2[64 * 64];
    __shared__ __align__(16) _Float16 Wt3[32 * 64];
    __shared__ float b1s[64], b2s[64], b3s[32];
    __shared__ __align__(16) _Float16 actA[4][16 * 64];  // per-wave ping
    __shared__ __align__(16) _Float16 actB[4][16 * 64];  // per-wave pong

    const int tid = threadIdx.x;

    // ---- cooperative weight conversion (fp32 -> f16, transpose, permute, pad)
    for (int i = tid; i < 64 * 64; i += 128) {
        int o = i >> 6, k = i & 63;
        float w1;
        if (k < 32)       w1 = W1[(k + F_DIM) * 64 + o];   // h_node rows
        else if (k < DIN) w1 = W1[(k - 32) * 64 + o];      // x rows
        else              w1 = 0.0f;                       // pad
        Wt1[o * 64 + k] = (_Float16)w1;
        Wt2[o * 64 + k] = (_Float16)W2[k * 64 + o];
    }
    for (int i = tid; i < 32 * 64; i += 128) {
        int o = i >> 6, k = i & 63;
        Wt3[o * 64 + k] = (_Float16)W3[k * 32 + o];
    }
    if (tid < 64) { b1s[tid] = b1[tid]; b2s[tid] = b2[tid]; }
    if (tid < 32) { b3s[tid] = b3[tid]; }

    const int wv   = tid >> 5;
    const int lane = tid & 31;
    const int col  = lane & 15;     // A-row / B-col / C-col index
    const int hl   = lane >> 4;     // lane half
    const int m0   = (blockIdx.x * 4 + wv) * 16;

    _Float16* a0 = actA[wv];
    _Float16* a1 = actB[wv];

    // ---- per-wave input tile, permuted K layout ----
    // h_node: dominant 256MB stream -> one float4 (global_load_b128) per lane
    // per iteration; dest 4 halves are 8B-aligned -> packed ds_store_b64.
    #pragma unroll
    for (int it = 0; it < 4; ++it) {
        int idx  = it * 32 + lane;      // float4 index within tile: 0..127
        int mm   = idx >> 3;            // node row (8 float4 per node)
        int q    = idx & 7;             // float4 slot within node
        int node = m0 + mm;
        float4 v = make_float4(0.f, 0.f, 0.f, 0.f);
        if (node < NUM_NODES)
            v = ((const float4*)h_node)[(long long)node * 8 + q];
        v4h h;
        h.x = (_Float16)v.x; h.y = (_Float16)v.y;
        h.z = (_Float16)v.z; h.w = (_Float16)v.w;
        *(v4h*)(a0 + mm * 64 + q * 4) = h;
    }
    // x (k=32..36) + zero pad (k=37..63)
    for (int t = lane; t < 16 * 32; t += 32) {
        int mm = t >> 5, k = 32 + (t & 31);
        int node = m0 + mm;
        float v = 0.0f;
        if (k < DIN && node < NUM_NODES) v = x[(long long)node * F_DIM + (k - 32)];
        a0[mm * 64 + k] = (_Float16)v;
    }
    __syncthreads();

    // ---- layer 1: [16x64] @ [64x64] + b1, ReLU -> a1 ----
    {
        v16h af0 = load_frag(a0 + col * 64, 0, hl);
        v16h af1 = load_frag(a0 + col * 64, 1, hl);
        #pragma unroll
        for (int n = 0; n < 4; ++n) {
            float bias = b1s[n * 16 + col];
            v8f c = {bias, bias, bias, bias, bias, bias, bias, bias};
            v16h bf0 = load_frag(Wt1 + (n * 16 + col) * 64, 0, hl);
            v16h bf1 = load_frag(Wt1 + (n * 16 + col) * 64, 1, hl);
            c = __builtin_amdgcn_wmma_f32_16x16x32_f16(false, af0, false, bf0,
                                                       (short)0, c, false, false);
            c = __builtin_amdgcn_wmma_f32_16x16x32_f16(false, af1, false, bf1,
                                                       (short)0, c, false, false);
            #pragma unroll
            for (int r = 0; r < 8; ++r)
                a1[(r + hl * 8) * 64 + n * 16 + col] = (_Float16)fmaxf(c[r], 0.0f);
        }
    }
    __syncthreads();

    // ---- layer 2: [16x64] @ [64x64] + b2, ReLU -> a0 ----
    {
        v16h af0 = load_frag(a1 + col * 64, 0, hl);
        v16h af1 = load_frag(a1 + col * 64, 1, hl);
        #pragma unroll
        for (int n = 0; n < 4; ++n) {
            float bias = b2s[n * 16 + col];
            v8f c = {bias, bias, bias, bias, bias, bias, bias, bias};
            v16h bf0 = load_frag(Wt2 + (n * 16 + col) * 64, 0, hl);
            v16h bf1 = load_frag(Wt2 + (n * 16 + col) * 64, 1, hl);
            c = __builtin_amdgcn_wmma_f32_16x16x32_f16(false, af0, false, bf0,
                                                       (short)0, c, false, false);
            c = __builtin_amdgcn_wmma_f32_16x16x32_f16(false, af1, false, bf1,
                                                       (short)0, c, false, false);
            #pragma unroll
            for (int r = 0; r < 8; ++r)
                a0[(r + hl * 8) * 64 + n * 16 + col] = (_Float16)fmaxf(c[r], 0.0f);
        }
    }
    __syncthreads();

    // ---- layer 3: [16x64] @ [64x32] + b3 -> registers ----
    v8f cout[2];
    {
        v16h af0 = load_frag(a0 + col * 64, 0, hl);
        v16h af1 = load_frag(a0 + col * 64, 1, hl);
        #pragma unroll
        for (int n = 0; n < 2; ++n) {
            float bias = b3s[n * 16 + col];
            v8f c = {bias, bias, bias, bias, bias, bias, bias, bias};
            v16h bf0 = load_frag(Wt3 + (n * 16 + col) * 64, 0, hl);
            v16h bf1 = load_frag(Wt3 + (n * 16 + col) * 64, 1, hl);
            c = __builtin_amdgcn_wmma_f32_16x16x32_f16(false, af0, false, bf0,
                                                       (short)0, c, false, false);
            c = __builtin_amdgcn_wmma_f32_16x16x32_f16(false, af1, false, bf1,
                                                       (short)0, c, false, false);
            cout[n] = c;
        }
    }

    // ---- segment-CSR sum: lane owns feature col (+16) for 8 consecutive rows
    // C layout: element r -> node row (r + hl*8). Run-length-compress rows in
    // the same segment before issuing global_atomic_add_f32.
    const int node0 = m0 + hl * 8;
    if (node0 < NUM_NODES) {
        int lo = 0, hi = NUM_DAGS;            // ptr[lo] <= node0 < ptr[hi]
        while (hi - lo > 1) {
            int mid = (lo + hi) >> 1;
            if (ptr[mid] <= node0) lo = mid; else hi = mid;
        }
        int seg = lo;
        int segEnd = ptr[seg + 1];
        float s0 = 0.0f, s1 = 0.0f;
        #pragma unroll
        for (int r = 0; r < 8; ++r) {
            int node = node0 + r;
            if (node >= NUM_NODES) break;
            if (node >= segEnd) {
                atomicAdd(out + (long long)seg * E_DIM + col,      s0);
                atomicAdd(out + (long long)seg * E_DIM + 16 + col, s1);
                s0 = 0.0f; s1 = 0.0f;
                do { ++seg; segEnd = ptr[seg + 1]; } while (node >= segEnd);
            }
            s0 += cout[0][r];
            s1 += cout[1][r];
        }
        atomicAdd(out + (long long)seg * E_DIM + col,      s0);
        atomicAdd(out + (long long)seg * E_DIM + 16 + col, s1);
    }
}

__global__ void zero_out_kernel(float* __restrict__ out, int n) {
    int i = blockIdx.x * blockDim.x + threadIdx.x;
    if (i < n) out[i] = 0.0f;
}

extern "C" void kernel_launch(void* const* d_in, const int* in_sizes, int n_in,
                              void* d_out, int out_size, void* d_ws, size_t ws_size,
                              hipStream_t stream) {
    const float* h_node = (const float*)d_in[0];
    const float* x      = (const float*)d_in[1];
    const float* W1     = (const float*)d_in[2];
    const float* b1     = (const float*)d_in[3];
    const float* W2     = (const float*)d_in[4];
    const float* b2     = (const float*)d_in[5];
    const float* W3     = (const float*)d_in[6];
    const float* b3     = (const float*)d_in[7];
    const int*   ptr    = (const int*)d_in[8];
    float* out = (float*)d_out;

    zero_out_kernel<<<(out_size + 255) / 256, 256, 0, stream>>>(out, out_size);

    const int nodes_per_block = 64;               // 4 waves x 16-node tiles
    const int blocks = (NUM_NODES + nodes_per_block - 1) / nodes_per_block;
    dag_mlp_segsum<<<blocks, 128, 0, stream>>>(h_node, x, W1, b1, W2, b2,
                                               W3, b3, ptr, out);
}
